// MultiheadAttention_44985487459042
// MI455X (gfx1250) — matrix operand order repaired and checked by reading, not compiled, not run on hardware
//
#include <hip/hip_runtime.h>
#include <hip/hip_bf16.h>

typedef __bf16 bf16_t;
typedef __attribute__((ext_vector_type(16))) __bf16 v16bf;
typedef __attribute__((ext_vector_type(8)))  float  v8f;
typedef __attribute__((ext_vector_type(4)))  int    v4i;

#define L_DIM   2048
#define S_DIM   2048
#define N_BATCH 2
#define D_DIM   1024
#define H_HEADS 16
#define HD_DIM  64
#define B_HEADS (N_BATCH * H_HEADS)   // 32

#if __has_builtin(__builtin_amdgcn_global_load_async_to_lds_b128)
#define HAS_ASYNC_LDS 1
#endif

// 16-byte global -> LDS copy. Async (ASYNCcnt-tracked, no VGPR round trip) when
// the gfx1250 builtin is available; synchronous fallback otherwise.
// Builtin prototype (from hipcc diagnostic): (v4i AS1*, v4i AS3*, imm, imm).
static __device__ __forceinline__ void async_copy16(void* lds_dst, const void* g_src) {
#if defined(HAS_ASYNC_LDS)
  __builtin_amdgcn_global_load_async_to_lds_b128(
      (__attribute__((address_space(1))) v4i*)g_src,
      (__attribute__((address_space(3))) v4i*)lds_dst, 0, 0);
#else
  *(float4*)lds_dst = *(const float4*)g_src;
#endif
}

template <int N>
static __device__ __forceinline__ void wait_async() {
#if __has_builtin(__builtin_amdgcn_s_wait_asynccnt)
  __builtin_amdgcn_s_wait_asynccnt(N);
#elif defined(HAS_ASYNC_LDS)
  asm volatile("s_wait_asynccnt %0" ::"n"(N));
#endif
}

static __device__ __forceinline__ bf16_t f2bf(float x) { return (bf16_t)x; }

// ---------------- WMMA fragment loaders (wave32, 16x16x32 bf16) -------------
// A fragment: 16(M) x 32(K). Lane = M (mod 16), lane group g = lane>>4.
// Element i: K = 16*(i/8) + 8*g + (i%8).   (ISA 7.12.2, 16-bit A 16x32)
static __device__ __forceinline__ v16bf load_a_f32(const float* __restrict__ p,
                                                   int ld, int lane) {
  const int m = lane & 15, g = lane >> 4;
  const float* row = p + (size_t)m * ld + (g << 3);
  v16bf a;
#pragma unroll
  for (int i = 0; i < 8; ++i) a[i] = f2bf(row[i]);          // K = 8g + i
#pragma unroll
  for (int i = 0; i < 8; ++i) a[i + 8] = f2bf(row[16 + i]); // K = 16 + 8g + i
  return a;
}

static __device__ __forceinline__ v16bf load_a_bf16(const bf16_t* __restrict__ p,
                                                    int ld, int lane) {
  const int m = lane & 15, g = lane >> 4;
  const bf16_t* row = p + (size_t)m * ld + (g << 3);
  v16bf a;
#pragma unroll
  for (int i = 0; i < 8; ++i) a[i] = row[i];
#pragma unroll
  for (int i = 0; i < 8; ++i) a[i + 8] = row[16 + i];
  return a;
}

// B fragment: 32(K) x 16(N), where B[k][n] = M[n][k] (rows of the matrix are
// the output columns: W^T for Y=X*W^T, K^T for Q*K^T). Lane = N (mod 16),
// group g: element i -> K = 16*(i/8) + 8g + (i%8).
static __device__ __forceinline__ v16bf load_bT_f32(const float* p, int ld, int lane) {
  const int n = lane & 15, g = lane >> 4;
  const float* row = p + (size_t)n * ld + (g << 3);
  v16bf b;
#pragma unroll
  for (int i = 0; i < 16; ++i) b[i] = f2bf(row[(i & 7) + ((i >> 3) << 4)]);
  return b;
}

static __device__ __forceinline__ v16bf load_bT_bf16(const bf16_t* p, int ld, int lane) {
  const int n = lane & 15, g = lane >> 4;
  const bf16_t* row = p + (size_t)n * ld + (g << 3);
  v16bf b;
#pragma unroll
  for (int i = 0; i < 16; ++i) b[i] = row[(i & 7) + ((i >> 3) << 4)];
  return b;
}

// B fragment from a genuinely row-major B (B[k][n] = M[k*ld + n]) — used for V.
static __device__ __forceinline__ v16bf load_b_rm_bf16(const bf16_t* p, int ld, int lane) {
  const int n = lane & 15, g = lane >> 4;
  const bf16_t* col = p + (size_t)(g << 3) * ld + n;
  v16bf b;
#pragma unroll
  for (int i = 0; i < 16; ++i) b[i] = col[(size_t)((i & 7) + ((i >> 3) << 4)) * ld];
  return b;
}

static __device__ __forceinline__ v8f wmma_bf16(v16bf a, v16bf b, v8f c) {
  // 8 args: (neg_a, A, neg_b, B, c_mod, C, reuse_a, reuse_b)
  return __builtin_amdgcn_wmma_f32_16x16x32_bf16(false, a, false, b,
                                                 (short)0, c, false, false);
}

// ------------------------- Kernel 1: QKV projection -------------------------
// grid: (32 row-blocks of 128, 16 col-blocks of 64, 3 slices q/k/v), block 256.
// All 8 waves share col0 -> stage the 64x32 f32 W slab in LDS (double-buffered,
// async) so each W element is fetched from global exactly once per block.
__global__ __launch_bounds__(256) void qkv_proj_kernel(
    const float* __restrict__ query, const float* __restrict__ key,
    const float* __restrict__ value, const float* __restrict__ w_in,
    const float* __restrict__ b_in, bf16_t* __restrict__ qb,
    bf16_t* __restrict__ kb, bf16_t* __restrict__ vb) {
  __shared__ float wslab[2][64 * 32];          // 2 x 8 KB
  const int tid = threadIdx.x;
  const int lane = tid & 31, wave = tid >> 5;
  const int p = blockIdx.z;
  const int row0 = blockIdx.x * 128 + wave * 16;
  const int col0 = blockIdx.y * 64;
  const float* src  = (p == 0) ? query : (p == 1) ? key : value;
  const float* W    = w_in + (size_t)p * D_DIM * D_DIM;
  const float* bias = b_in + p * D_DIM;
  bf16_t* dst = (p == 0) ? qb : (p == 1) ? kb : vb;
  const float scale = (p == 0) ? 0.125f : 1.0f;  // HD^-0.5 = 1/8 on q only

  auto stage = [&](int buf, int k0) {            // 64 rows x 32 f32 = 512 chunks
#pragma unroll
    for (int i = 0; i < 2; ++i) {
      const int c = tid + 256 * i;               // 0..511
      const int r = c >> 3;                      // output column within block
      const int o = (c & 7) * 4;                 // float offset within row
      async_copy16(&wslab[buf][r * 32 + o],
                   W + (size_t)(col0 + r) * D_DIM + k0 + o);
    }
  };

  stage(0, 0);
  v8f acc[4] = {{}, {}, {}, {}};
  for (int it = 0; it < D_DIM / 32; ++it) {
    const int k0 = it * 32;
    const int cur = it & 1;
    if (it + 1 < D_DIM / 32) {
      stage(cur ^ 1, k0 + 32);   // prefetch next slab
      wait_async<2>();           // current slab (issued earlier) must be done
    } else {
      wait_async<0>();
    }
    __syncthreads();
    v16bf a = load_a_f32(src + (size_t)row0 * D_DIM + k0, D_DIM, lane);
#pragma unroll
    for (int j = 0; j < 4; ++j) {
      v16bf bfr = load_bT_f32(&wslab[cur][(16 * j) * 32], 32, lane);
      acc[j] = wmma_bf16(a, bfr, acc[j]);
    }
    __syncthreads();             // protect slab before it gets overwritten
  }

  const int g = lane >> 4, nn = lane & 15;
#pragma unroll
  for (int j = 0; j < 4; ++j) {
#pragma unroll
    for (int r = 0; r < 8; ++r) {
      const int m = row0 + g * 8 + r;       // m = l*N + n  (row of (L,N,D) flat)
      const int d = col0 + 16 * j + nn;     // output feature
      const float v = (acc[j][r] + bias[d]) * scale;
      const int l = m >> 1, nb = m & 1;     // N_BATCH == 2
      const int h = d >> 6, hd = d & 63;    // HD == 64
      const int bh = nb * H_HEADS + h;      // head-batch index
      dst[((size_t)bh * S_DIM + l) * HD_DIM + hd] = f2bf(v);
    }
  }
}

// ------------------------- Kernel 2: S = Q K^T ------------------------------
// grid: (16, 128, 32); block 256. Stage the shared 16x64 Q tile (was loaded
// 8x redundantly) and the 128x64 K block in LDS with one bulk async copy.
__global__ __launch_bounds__(256) void qk_scores_kernel(
    const bf16_t* __restrict__ qb, const bf16_t* __restrict__ kb,
    float* __restrict__ out_w) {
  __shared__ bf16_t qtile[16 * HD_DIM];    // 2 KB
  __shared__ bf16_t ktile[128 * HD_DIM];   // 16 KB
  const int tid = threadIdx.x;
  const int lane = tid & 31, wave = tid >> 5;
  const int b   = blockIdx.z;
  const int l0  = blockIdx.y * 16;
  const int s0b = blockIdx.x * 128;
  const bf16_t* qsrc = qb + ((size_t)b * L_DIM + l0) * HD_DIM;  // 2 KB contig
  const bf16_t* ksrc = kb + ((size_t)b * S_DIM + s0b) * HD_DIM; // 16 KB contig

  if (tid < 128) async_copy16(&qtile[tid * 8], qsrc + tid * 8);
#pragma unroll
  for (int i = 0; i < 4; ++i) {
    const int c = tid + 256 * i;
    async_copy16(&ktile[c * 8], ksrc + c * 8);
  }
  wait_async<0>();
  __syncthreads();

  const bf16_t* kwave = &ktile[wave * 16 * HD_DIM];
  v8f acc = {};
#pragma unroll
  for (int k0 = 0; k0 < HD_DIM; k0 += 32) {
    v16bf a  = load_a_bf16(qtile + k0, HD_DIM, lane);
    v16bf bf = load_bT_bf16(kwave + k0, HD_DIM, lane);  // B[k][n] = K[s0+n][k]
    acc = wmma_bf16(a, bf, acc);
  }
  const int g = lane >> 4, nn = lane & 15;
  const int s0 = s0b + wave * 16;
  float* orow = out_w + ((size_t)b * L_DIM + l0) * S_DIM + s0;
#pragma unroll
  for (int r = 0; r < 8; ++r)
    orow[(size_t)(g * 8 + r) * S_DIM + nn] = acc[r];
}

// ------------------------- Kernel 3: row softmax ----------------------------
// grid: B*L rows, block 256, 8 elements per thread; in-place on attn_weights.
__global__ __launch_bounds__(256) void softmax_kernel(float* __restrict__ w) {
  __shared__ float red[256];
  float* p = w + (size_t)blockIdx.x * S_DIM;
  const int tid = threadIdx.x;
  float vals[8];
  float m = -1e30f;
#pragma unroll
  for (int i = 0; i < 8; ++i) { vals[i] = p[tid + i * 256]; m = fmaxf(m, vals[i]); }
  red[tid] = m; __syncthreads();
  for (int off = 128; off > 0; off >>= 1) {
    if (tid < off) red[tid] = fmaxf(red[tid], red[tid + off]);
    __syncthreads();
  }
  m = red[0]; __syncthreads();
  float s = 0.f;
#pragma unroll
  for (int i = 0; i < 8; ++i) { vals[i] = __expf(vals[i] - m); s += vals[i]; }
  red[tid] = s; __syncthreads();
  for (int off = 128; off > 0; off >>= 1) {
    if (tid < off) red[tid] += red[tid + off];
    __syncthreads();
  }
  const float inv = 1.0f / red[0];
#pragma unroll
  for (int i = 0; i < 8; ++i) p[tid + i * 256] = vals[i] * inv;
}

// ------------------------- Kernel 4: O = P V --------------------------------
// grid: (16, 32); block 256; wave = 16 rows x full HD(64), K = S = 2048.
// All 8 waves share head b -> stage the 32x64 bf16 V slab (4 KB, contiguous)
// in LDS, double-buffered with one async b128 per thread per iteration.
__global__ __launch_bounds__(256) void pv_kernel(
    const float* __restrict__ w, const bf16_t* __restrict__ vb,
    bf16_t* __restrict__ ctx) {
  __shared__ bf16_t vtile[2][32 * HD_DIM];   // 2 x 4 KB
  const int tid = threadIdx.x;
  const int lane = tid & 31, wave = tid >> 5;
  const int b  = blockIdx.y;
  const int l0 = (blockIdx.x * 8 + wave) * 16;
  const float*  pbase = w  + ((size_t)b * L_DIM + l0) * S_DIM;
  const bf16_t* vslab = vb + (size_t)b * S_DIM * HD_DIM;  // rows contiguous

  async_copy16(&vtile[0][tid * 8], vslab + tid * 8);       // k0 = 0 slab
  v8f acc[4] = {{}, {}, {}, {}};
  for (int it = 0; it < S_DIM / 32; ++it) {
    const int k0 = it * 32;
    const int cur = it & 1;
    if (it + 1 < S_DIM / 32) {
      async_copy16(&vtile[cur ^ 1][tid * 8],
                   vslab + (size_t)(k0 + 32) * HD_DIM + tid * 8);
      wait_async<1>();
    } else {
      wait_async<0>();
    }
    __syncthreads();
    v16bf a = load_a_f32(pbase + k0, S_DIM, lane);  // P read exactly once
#pragma unroll
    for (int j = 0; j < 4; ++j) {
      v16bf bf = load_b_rm_bf16(&vtile[cur][16 * j], HD_DIM, lane);
      acc[j] = wmma_bf16(a, bf, acc[j]);
    }
    __syncthreads();
  }
  const int g = lane >> 4, nn = lane & 15;
  const int nb = b >> 4, h = b & 15;   // b = nb*H + h
#pragma unroll
  for (int j = 0; j < 4; ++j) {
#pragma unroll
    for (int r = 0; r < 8; ++r) {
      const int l = l0 + g * 8 + r;
      const int col = h * HD_DIM + 16 * j + nn;
      ctx[((size_t)l * N_BATCH + nb) * D_DIM + col] = f2bf(acc[j][r]);
    }
  }
}

// ------------------------- Kernel 5: output projection ----------------------
// grid: (32, 16); block 256; wave = 16x64 tile, K = 1024. Same LDS-staged
// W-slab scheme as the QKV projection.
__global__ __launch_bounds__(256) void out_proj_kernel(
    const bf16_t* __restrict__ ctx, const float* __restrict__ wo,
    const float* __restrict__ bo, float* __restrict__ out) {
  __shared__ float wslab[2][64 * 32];          // 2 x 8 KB
  const int tid = threadIdx.x;
  const int lane = tid & 31, wave = tid >> 5;
  const int row0 = blockIdx.x * 128 + wave * 16;
  const int col0 = blockIdx.y * 64;

  auto stage = [&](int buf, int k0) {
#pragma unroll
    for (int i = 0; i < 2; ++i) {
      const int c = tid + 256 * i;
      const int r = c >> 3;
      const int o = (c & 7) * 4;
      async_copy16(&wslab[buf][r * 32 + o],
                   wo + (size_t)(col0 + r) * D_DIM + k0 + o);
    }
  };

  stage(0, 0);
  v8f acc[4] = {{}, {}, {}, {}};
  for (int it = 0; it < D_DIM / 32; ++it) {
    const int k0 = it * 32;
    const int cur = it & 1;
    if (it + 1 < D_DIM / 32) {
      stage(cur ^ 1, k0 + 32);
      wait_async<2>();
    } else {
      wait_async<0>();
    }
    __syncthreads();
    v16bf a = load_a_bf16(ctx + (size_t)row0 * D_DIM + k0, D_DIM, lane);
#pragma unroll
    for (int j = 0; j < 4; ++j) {
      v16bf bfr = load_bT_f32(&wslab[cur][(16 * j) * 32], 32, lane);
      acc[j] = wmma_bf16(a, bfr, acc[j]);
    }
    __syncthreads();
  }
  const int g = lane >> 4, nn = lane & 15;
#pragma unroll
  for (int j = 0; j < 4; ++j) {
#pragma unroll
    for (int r = 0; r < 8; ++r) {
      const int m = row0 + g * 8 + r;
      const int d = col0 + 16 * j + nn;
      out[(size_t)m * D_DIM + d] = acc[j][r] + bo[d];
    }
  }
}

// ---------------------------------------------------------------------------
extern "C" void kernel_launch(void* const* d_in, const int* in_sizes, int n_in,
                              void* d_out, int out_size, void* d_ws, size_t ws_size,
                              hipStream_t stream) {
  const float* query = (const float*)d_in[0];
  const float* key   = (const float*)d_in[1];
  const float* value = (const float*)d_in[2];
  const float* w_in  = (const float*)d_in[3];
  const float* b_in  = (const float*)d_in[4];
  const float* wo    = (const float*)d_in[5];
  const float* bo    = (const float*)d_in[6];

  float* out_attn = (float*)d_out;                                   // (L,N,D)
  float* out_w    = (float*)d_out + (size_t)L_DIM * N_BATCH * D_DIM; // (B,L,S)

  char* ws = (char*)d_ws;
  const size_t SZ_HEADS = (size_t)B_HEADS * S_DIM * HD_DIM * sizeof(bf16_t); // 8 MB
  bf16_t* qb  = (bf16_t*)(ws);
  bf16_t* kb  = (bf16_t*)(ws + SZ_HEADS);
  bf16_t* vb  = (bf16_t*)(ws + 2 * SZ_HEADS);
  bf16_t* ctx = (bf16_t*)(ws + 3 * SZ_HEADS);                        // 8 MB

  qkv_proj_kernel<<<dim3(32, 16, 3), 256, 0, stream>>>(query, key, value,
                                                       w_in, b_in, qb, kb, vb);
  qk_scores_kernel<<<dim3(16, 128, 32), 256, 0, stream>>>(qb, kb, out_w);
  softmax_kernel<<<dim3(B_HEADS * L_DIM), 256, 0, stream>>>(out_w);
  pv_kernel<<<dim3(16, 32), 256, 0, stream>>>(out_w, vb, ctx);
  out_proj_kernel<<<dim3(32, 16), 256, 0, stream>>>(ctx, wo, bo, out_attn);
}
